// GNNWithAttention_78623671321105
// MI455X (gfx1250) — matrix operand order repaired and checked by reading, not compiled, not run on hardware
//
#include <hip/hip_runtime.h>
#include <hip/hip_bf16.h>

// GAT (2x GATConv, 4 heads, concat=False) + global mean pool + FC for gfx1250.
// GEMMs: v_wmma_f32_16x16x32_f16 with pre-packed f16 operands (B pre-swizzled in
// global, A packed to f16 in LDS), f32 accumulate. Edge phase fp32 + prefetch.

#define IN_DIM 128
#define HID    64
#define HEADS  4
#define OUTD   16

typedef __attribute__((ext_vector_type(16))) _Float16     v16h;
typedef __attribute__((ext_vector_type(2)))  _Float16     v2h;
typedef __attribute__((ext_vector_type(8)))  float        v8f;
typedef __attribute__((ext_vector_type(8)))  unsigned int v8u;

// ---------------------------------------------------------------- utilities

__global__ void k_fill(float* __restrict__ p, float v, int n) {
  int i = blockIdx.x * blockDim.x + threadIdx.x;
  if (i < n) p[i] = v;
}

__device__ __forceinline__ void atomicMaxF(float* addr, float val) {
  int old = __float_as_int(*addr);
  while (__int_as_float(old) < val) {
    int assumed = old;
    old = atomicCAS((int*)addr, assumed, __float_as_int(val));
    if (old == assumed) break;
  }
}

__device__ __forceinline__ void edge_sd(const long long* __restrict__ ei,
                                        int E, int e, int& s, int& d) {
  if (e < E) { s = (int)ei[e]; d = (int)ei[(size_t)E + e]; }
  else       { s = e - E; d = s; }          // appended self-loops
}

__device__ __forceinline__ unsigned int pack2h(float a, float b) {
  v2h p; p[0] = (_Float16)a; p[1] = (_Float16)b;
  return __builtin_bit_cast(unsigned int, p);
}

// -------------------------------------------------- W pre-pack (f16, B order)
// Wh dword index: (((nt*KS + ksi)*32 + lane)*8 + v)
//   lane(0-15)=col within tile, K = ksi*32 + (lane/16)*16 + v*2 (+0,+1)
template<int K>
__global__ void k_convW(const float* __restrict__ W, unsigned int* __restrict__ Wh) {
  constexpr int KS = K / 32;
  int t = blockIdx.x * blockDim.x + threadIdx.x;
  if (t >= 16 * KS * 32 * 8) return;
  int v    = t & 7;
  int lane = (t >> 3) & 31;
  int rest = t >> 8;                 // nt*KS + ksi
  int ksi  = rest % KS;
  int nt   = rest / KS;
  int col  = nt * 16 + (lane & 15);
  int k    = ksi * 32 + ((lane >> 4) << 4) + v * 2;
  Wh[t] = pack2h(W[k * 256 + col], W[(k + 1) * 256 + col]);
}

// ---------------------------------------------------------------- WMMA GEMM
// H[M,256] = A[M,K] @ W[K,256].  grid.x = M/16, block = 128 (4 waves).
// Wave w computes the 16x64 slab for head w (4 16x16 tiles).
template<int K>
__global__ __launch_bounds__(128)
void k_gemm(const float* __restrict__ A, const unsigned int* __restrict__ Wh,
            float* __restrict__ H) {
  constexpr int KS = K / 32;
  constexpr int KW = K / 2;                 // packed dwords per A row
  __shared__ unsigned int As[16 * KW];      // <= 4 KB, f16 pairs
  const int mt   = blockIdx.x;
  const int tid  = threadIdx.x;
  const int wave = tid >> 5;                // 0..3 == head id
  const int lane = tid & 31;
  const int lmod = lane & 15;
  const int lhi  = lane >> 4;               // 0/1

  // stage + convert A strip (coalesced float2 loads, packed f16 stores)
  const float2* Ab = (const float2*)(A + (size_t)mt * 16 * K);
  for (int i = tid; i < 16 * KW; i += 128) {
    float2 ab = Ab[i];
    As[i] = pack2h(ab.x, ab.y);
  }
  __syncthreads();

  v8f acc[4] = {};
#pragma unroll
  for (int ksi = 0; ksi < KS; ++ksi) {
    // A fragment: dwords {ksi*16 + lhi*8 + 0..7} of row lmod
    const unsigned int* ar = &As[lmod * KW + ksi * 16];
    uint4 lo = *(const uint4*)(ar + lhi * 8);
    uint4 hi = *(const uint4*)(ar + lhi * 8 + 4);
    v8u au; au[0]=lo.x; au[1]=lo.y; au[2]=lo.z; au[3]=lo.w;
            au[4]=hi.x; au[5]=hi.y; au[6]=hi.z; au[7]=hi.w;
    v16h af = __builtin_bit_cast(v16h, au);
#pragma unroll
    for (int t = 0; t < 4; ++t) {
      const int nt = wave * 4 + t;
      v16h bf = __builtin_bit_cast(
          v16h, *(const v8u*)&Wh[(((size_t)nt * KS + ksi) * 32 + lane) * 8]);
      acc[t] = __builtin_amdgcn_wmma_f32_16x16x32_f16(
          false, af, false, bf, (short)0, acc[t], false, false);
    }
  }
  // C/D f32 16x16 layout: vgpr v -> row v (+8 for lanes 16-31), col = lane%16
#pragma unroll
  for (int t = 0; t < 4; ++t) {
    const int ncol = (wave * 4 + t) * 16 + lmod;
#pragma unroll
    for (int v = 0; v < 8; ++v) {
      int r = mt * 16 + v + (lhi << 3);
      H[(size_t)r * 256 + ncol] = acc[t][v];
    }
  }
}

// ------------------------------------------------------- attention pipeline

// alpha_s[n,h] = sum_d h[n,h*64+d]*a_src[h*64+d]   (and same for dst)
__global__ void k_alpha(const float* __restrict__ h,
                        const float* __restrict__ a_src,
                        const float* __restrict__ a_dst,
                        float* __restrict__ as_, float* __restrict__ ad_, int n_nodes) {
  int t = blockIdx.x * blockDim.x + threadIdx.x;
  if (t >= n_nodes * HEADS) return;
  int n = t >> 2, hd = t & 3;
  const float* hp = h + (size_t)n * 256 + hd * HID;
  const float* s  = a_src + hd * HID;
  const float* d  = a_dst + hd * HID;
  float aS = 0.f, aD = 0.f;
#pragma unroll 8
  for (int k = 0; k < HID; ++k) { float v = hp[k]; aS += v * s[k]; aD += v * d[k]; }
  as_[t] = aS; ad_[t] = aD;
}

__device__ __forceinline__ float leaky(float x) { return x > 0.f ? x : 0.2f * x; }

__global__ void k_edge_max(const long long* __restrict__ ei, int E, int Etot,
                           const float* __restrict__ as_, const float* __restrict__ ad_,
                           float* __restrict__ emax) {
  int e = blockIdx.x * blockDim.x + threadIdx.x;
  if (e >= Etot) return;
  int s, d; edge_sd(ei, E, e, s, d);
#pragma unroll
  for (int h = 0; h < HEADS; ++h) {
    float v = leaky(as_[s * HEADS + h] + ad_[d * HEADS + h]);
    atomicMaxF(&emax[d * HEADS + h], v);
  }
}

__global__ void k_edge_expsum(const long long* __restrict__ ei, int E, int Etot,
                              const float* __restrict__ as_, const float* __restrict__ ad_,
                              const float* __restrict__ emax,
                              float4* __restrict__ exbuf, float* __restrict__ denom) {
  int e = blockIdx.x * blockDim.x + threadIdx.x;
  if (e >= Etot) return;
  int s, d; edge_sd(ei, E, e, s, d);
  float ex[HEADS];
#pragma unroll
  for (int h = 0; h < HEADS; ++h) {
    float v = leaky(as_[s * HEADS + h] + ad_[d * HEADS + h]);
    ex[h] = __expf(v - emax[d * HEADS + h]);
    atomicAdd(&denom[d * HEADS + h], ex[h]);
  }
  exbuf[e] = make_float4(ex[0], ex[1], ex[2], ex[3]);
}

// grid-stride over edges; 256 lanes cover (head,dim); prefetch next src row.
__global__ __launch_bounds__(256)
void k_edge_scatter(const long long* __restrict__ ei, int E, int Etot,
                    const float* __restrict__ h1,
                    const float* __restrict__ exbuf, const float* __restrict__ denom,
                    float* __restrict__ out) {
  const int t  = threadIdx.x;               // t = head*64 + dim
  const int hd = t >> 6;
  const int stride = gridDim.x;
  for (int e = blockIdx.x; e < Etot; e += stride) {
    int s, d; edge_sd(ei, E, e, s, d);
    int en = e + stride;
    if (en < Etot && (t & 15) == 0) {       // prefetch next source row (1 KB)
      int sn = (en < E) ? (int)ei[en] : (en - E);
      __builtin_prefetch(&h1[(size_t)sn * 256 + t], 0, 0);
    }
    float alpha = exbuf[(size_t)e * HEADS + hd] / denom[d * HEADS + hd];
    float m = h1[(size_t)s * 256 + t] * alpha;
    atomicAdd(&out[(size_t)d * 256 + t], m);
  }
}

// head mean + bias + ReLU  -> hrelu[n,64]
__global__ void k_head_mean(const float* __restrict__ out, const float* __restrict__ b,
                            float* __restrict__ hrelu, int n_nodes) {
  int t = blockIdx.x * blockDim.x + threadIdx.x;
  if (t >= n_nodes * HID) return;
  int n = t >> 6, d = t & 63;
  const float* o = out + (size_t)n * 256;
  float s = o[d] + o[64 + d] + o[128 + d] + o[192 + d];
  float v = s * 0.25f + b[d];
  hrelu[t] = v > 0.f ? v : 0.f;
}

// -------------------------------------------------------------- pool and FC

__global__ void k_pool(const float* __restrict__ hr, const long long* __restrict__ batch,
                       float* __restrict__ pooled, float* __restrict__ counts, int n_nodes) {
  int t = blockIdx.x * blockDim.x + threadIdx.x;
  if (t >= n_nodes * HID) return;
  int n = t >> 6, d = t & 63;
  int g = (int)batch[n];
  atomicAdd(&pooled[g * HID + d], hr[t]);
  if (d == 0) atomicAdd(&counts[g], 1.0f);
}

__global__ void k_fc(const float* __restrict__ pooled, const float* __restrict__ counts,
                     const float* __restrict__ fcW, const float* __restrict__ fcb,
                     float* __restrict__ y, int n_graphs) {
  int t = blockIdx.x * blockDim.x + threadIdx.x;
  if (t >= n_graphs * OUTD) return;
  int g = t >> 4, o = t & 15;
  float c = counts[g]; if (c < 1.f) c = 1.f;
  float acc = 0.f;
#pragma unroll 8
  for (int d = 0; d < HID; ++d) acc += pooled[g * HID + d] * fcW[d * OUTD + o];
  y[t] = acc / c + fcb[o];
}

// ------------------------------------------------------------------- launch

static inline int cdiv(int a, int b) { return (a + b - 1) / b; }

extern "C" void kernel_launch(void* const* d_in, const int* in_sizes, int n_in,
                              void* d_out, int out_size, void* d_ws, size_t ws_size,
                              hipStream_t stream) {
  const float*     x      = (const float*)d_in[0];
  const long long* ei     = (const long long*)d_in[1];
  const long long* batch  = (const long long*)d_in[2];
  const float*     W1     = (const float*)d_in[3];
  const float*     a1s    = (const float*)d_in[4];
  const float*     a1d    = (const float*)d_in[5];
  const float*     b1     = (const float*)d_in[6];
  const float*     W2     = (const float*)d_in[7];
  const float*     a2s    = (const float*)d_in[8];
  const float*     a2d    = (const float*)d_in[9];
  const float*     b2     = (const float*)d_in[10];
  const float*     fcW    = (const float*)d_in[11];
  const float*     fcb    = (const float*)d_in[12];
  float*           y      = (float*)d_out;

  const int N    = in_sizes[0] / IN_DIM;     // 50000 (divisible by 16)
  const int E    = in_sizes[1] / 2;          // 800000
  const int Etot = E + N;                    // + self loops
  const int G    = out_size / OUTD;          // 500

  // workspace layout; packed-f16 weights first (32B aligned), then floats
  unsigned int* wh1 = (unsigned int*)d_ws;                 // 128*256/2 = 16384
  unsigned int* wh2 = wh1 + 16384;                         //  64*256/2 =  8192
  float* ws = (float*)(wh2 + 8192);
  size_t off = 0;
  float* h      = ws + off; off += (size_t)N * 256;
  float* outb   = ws + off; off += (size_t)N * 256;
  float* as_    = ws + off; off += (size_t)N * HEADS;
  float* ad_    = ws + off; off += (size_t)N * HEADS;
  float* emax   = ws + off; off += (size_t)N * HEADS;
  float* denom  = ws + off; off += (size_t)N * HEADS;
  float* exbuf  = ws + off; off += (size_t)Etot * HEADS;
  float* hrelu1 = ws + off; off += (size_t)N * HID;
  float* hrelu2 = ws + off; off += (size_t)N * HID;
  float* pooled = ws + off; off += (size_t)G * HID;
  float* counts = ws + off; off += (size_t)G;
  (void)ws_size; (void)n_in;

  const int TB = 256;
  const int SCAT_BLOCKS = 32768;

  // ---------------- layer 1 ----------------
  k_convW<IN_DIM><<<cdiv(16384, TB), TB, 0, stream>>>(W1, wh1);
  k_fill<<<cdiv(N * 256, TB), TB, 0, stream>>>(outb, 0.f, N * 256);
  k_fill<<<cdiv(N * HEADS, TB), TB, 0, stream>>>(emax, -__builtin_huge_valf(), N * HEADS);
  k_fill<<<cdiv(N * HEADS, TB), TB, 0, stream>>>(denom, 0.f, N * HEADS);

  k_gemm<IN_DIM><<<N / 16, 128, 0, stream>>>(x, wh1, h);
  k_alpha<<<cdiv(N * HEADS, TB), TB, 0, stream>>>(h, a1s, a1d, as_, ad_, N);
  k_edge_max<<<cdiv(Etot, TB), TB, 0, stream>>>(ei, E, Etot, as_, ad_, emax);
  k_edge_expsum<<<cdiv(Etot, TB), TB, 0, stream>>>(ei, E, Etot, as_, ad_, emax,
                                                  (float4*)exbuf, denom);
  k_edge_scatter<<<SCAT_BLOCKS, 256, 0, stream>>>(ei, E, Etot, h, exbuf, denom, outb);
  k_head_mean<<<cdiv(N * HID, TB), TB, 0, stream>>>(outb, b1, hrelu1, N);

  // ---------------- layer 2 ----------------
  k_convW<HID><<<cdiv(8192, TB), TB, 0, stream>>>(W2, wh2);
  k_fill<<<cdiv(N * 256, TB), TB, 0, stream>>>(outb, 0.f, N * 256);
  k_fill<<<cdiv(N * HEADS, TB), TB, 0, stream>>>(emax, -__builtin_huge_valf(), N * HEADS);
  k_fill<<<cdiv(N * HEADS, TB), TB, 0, stream>>>(denom, 0.f, N * HEADS);

  k_gemm<HID><<<N / 16, 128, 0, stream>>>(hrelu1, wh2, h);
  k_alpha<<<cdiv(N * HEADS, TB), TB, 0, stream>>>(h, a2s, a2d, as_, ad_, N);
  k_edge_max<<<cdiv(Etot, TB), TB, 0, stream>>>(ei, E, Etot, as_, ad_, emax);
  k_edge_expsum<<<cdiv(Etot, TB), TB, 0, stream>>>(ei, E, Etot, as_, ad_, emax,
                                                  (float4*)exbuf, denom);
  k_edge_scatter<<<SCAT_BLOCKS, 256, 0, stream>>>(ei, E, Etot, h, exbuf, denom, outb);
  k_head_mean<<<cdiv(N * HID, TB), TB, 0, stream>>>(outb, b2, hrelu2, N);

  // ---------------- pool + fc ----------------
  k_fill<<<cdiv(G * HID, TB), TB, 0, stream>>>(pooled, 0.f, G * HID);
  k_fill<<<cdiv(G, TB), TB, 0, stream>>>(counts, 0.f, G);
  k_pool<<<cdiv(N * HID, TB), TB, 0, stream>>>(hrelu2, batch, pooled, counts, N);
  k_fc<<<cdiv(G * OUTD, TB), TB, 0, stream>>>(pooled, counts, fcW, fcb, y, G);
}
